// InteractionModule_12232066859521
// MI455X (gfx1250) — matrix operand order repaired and checked by reading, not compiled, not run on hardware
//
#include <hip/hip_runtime.h>

// ======================= CDNA5 (gfx1250) bf16 WMMA pipeline =======================
// Wave32-only. Matrix ops: v_wmma_f32_16x16x32_bf16 (f32 accumulation).
// Data movement: async global->LDS copies (ASYNCcnt) + double-buffered LDS pipeline.

typedef __bf16 bf16_t;
typedef __attribute__((ext_vector_type(16))) __bf16 v16bf;
typedef __attribute__((ext_vector_type(8)))  __bf16 bh8;
typedef __attribute__((ext_vector_type(8)))  float  v8f;

static constexpr int BATCH = 32;
static constexpr int LSEQ  = 1024;
static constexpr int DIM   = 1024;
// block tile: 256 threads = 8 waves; each wave computes 32 rows x 64 cols
static constexpr int BM = 256, BN = 64, BK = 32;
static constexpr float NEGC = -1.0e9f;

__device__ __forceinline__ bf16_t f32_to_bf16(float f) {
  unsigned int u = __builtin_bit_cast(unsigned int, f);
  u += 0x7FFFu + ((u >> 16) & 1u);                 // round-to-nearest-even
  unsigned short h = (unsigned short)(u >> 16);
  return __builtin_bit_cast(bf16_t, h);
}

struct __align__(16) GemmSmem {
  bf16_t At[2][BM][BK];   // double-buffered A tile [m][k]  (2 x 16KB)
  bf16_t Bt[2][BN][BK];   // double-buffered B tile [n][k]  (2 x 4KB)
};
static constexpr int ABUF = BM * BK;   // elements per A buffer
static constexpr int BBUF = BN * BK;   // elements per B buffer

// ---- CDNA5 async global->LDS copy (16B per lane), tracked by ASYNCcnt ----
__device__ __forceinline__ void async_cp16(void* lds_dst, const void* gsrc) {
  asm volatile("global_load_async_to_lds_b128 %0, %1, off"
               :: "v"((unsigned int)(unsigned long long)(uintptr_t)lds_dst),
                  "v"((unsigned long long)(uintptr_t)gsrc)
               : "memory");
}
__device__ __forceinline__ void wait_async_all() {
  asm volatile("s_wait_asynccnt 0x0" ::: "memory");
}

__device__ __forceinline__ void zero_acc(v8f acc[2][4]) {
#pragma unroll
  for (int m = 0; m < 2; ++m)
#pragma unroll
    for (int i = 0; i < 4; ++i)
#pragma unroll
      for (int j = 0; j < 8; ++j) acc[m][i][j] = 0.0f;
}

// stage A tile (BM x BK = 16KB): 4 async 16B copies per thread
__device__ __forceinline__ void stage_a(GemmSmem* sm, int buf,
                                        const bf16_t* __restrict__ A, int lda, int k0) {
  const int tid = threadIdx.x;
#pragma unroll
  for (int c = 0; c < 4; ++c) {
    int idx = tid + c * 256;            // 0..1023 chunks of 16B
    int row = idx >> 2;
    int kc  = (idx & 3) * 8;
    async_cp16(&sm->At[buf][row][kc], &A[(long long)row * lda + k0 + kc]);
  }
}

// stage B tile (BN x BK = 4KB), stored [n][k].
// BT=true : B is [N,K] row-major -> direct async copy
// BT=false: B is [K,N] row-major -> synchronous transpose into LDS
template<bool BT>
__device__ __forceinline__ void stage_b(GemmSmem* sm, int buf,
                                        const bf16_t* __restrict__ Bm, int ldb, int k0) {
  const int tid = threadIdx.x;
  if (BT) {
    int row = tid >> 2;                 // n
    int kc  = (tid & 3) * 8;
    async_cp16(&sm->Bt[buf][row][kc], &Bm[(long long)row * ldb + k0 + kc]);
  } else {
    int kr = tid >> 3;                  // k within tile
    int nc = (tid & 7) * 8;
    bh8 vals = *(const bh8*)&Bm[(long long)(k0 + kr) * ldb + nc];
#pragma unroll
    for (int j = 0; j < 8; ++j) sm->Bt[buf][nc + j][kr] = vals[j];
  }
}

// acc += A(BM x K) * op(B); A row-major (lda), pre-offset to tile origin.
// BT=true : B is [N,K] row-major (computes X @ B^T), pre-offset to row n0.
// BT=false: B is [K,N] row-major,                    pre-offset to col n0.
// Fragment gathers follow CDNA5 ISA 7.12.2 bf16 layouts (two ds_load_b128 each).
// Pipeline: stage s+1 copies are issued right after barrier(s); all reads of the
// buffer they overwrite were dscnt-waited before any wave signaled barrier(s).
// The stage loop is kept rolled so each accumulator has a single WMMA def per
// iteration and stays pinned in its register tuple (no cross-tuple v_mov churn).
template<bool BT>
__device__ inline void gemm_core(const bf16_t* __restrict__ A, int lda,
                                 const bf16_t* __restrict__ Bm, int ldb,
                                 int K, GemmSmem* sm, v8f acc[2][4]) {
  const int tid  = threadIdx.x;
  const int wave = tid >> 5;
  const int lane = tid & 31;
  const int l15  = lane & 15;
  const int akb  = (lane >> 4) * 8;
  const int bkb  = (lane >> 4) * 8;
  const int nstages = K / BK;

  // per-wave fragment base pointers (buffer 0); buffer select = + ABUF/BBUF
  const int ar0 = wave * 32 + l15;
  const bf16_t* a0p = &sm->At[0][ar0][akb];
  const bf16_t* a1p = &sm->At[0][ar0 + 16][akb];
  const bf16_t* bp  = &sm->Bt[0][l15][bkb];

  __syncthreads();                       // protect prologue writes vs prior LDS readers
  stage_a(sm, 0, A, lda, 0);
  stage_b<BT>(sm, 0, Bm, ldb, 0);

#pragma clang loop unroll(disable)
  for (int s = 0; s < nstages; ++s) {
    const int cur = s & 1;
    wait_async_all();                    // stage-s copies (this thread) complete
    __syncthreads();                     // => everyone's stage-s data visible in LDS
    if (s + 1 < nstages) {               // overlap next-stage copies with WMMA work
      stage_a(sm, cur ^ 1, A, lda, (s + 1) * BK);
      stage_b<BT>(sm, cur ^ 1, Bm, ldb, (s + 1) * BK);
    }

    // ---- A fragments: rows wave*32 + {l15, l15+16} ----
    const bf16_t* ac0 = a0p + cur * ABUF;
    const bf16_t* ac1 = a1p + cur * ABUF;
    const bf16_t* bc  = bp  + cur * BBUF;
    bh8 a0lo = *(const bh8*)(ac0);
    bh8 a0hi = *(const bh8*)(ac0 + 16);
    bh8 a1lo = *(const bh8*)(ac1);
    bh8 a1hi = *(const bh8*)(ac1 + 16);
    v16bf a0, a1;
#pragma unroll
    for (int j = 0; j < 8; ++j) {
      a0[j] = a0lo[j]; a0[j + 8] = a0hi[j];
      a1[j] = a1lo[j]; a1[j + 8] = a1hi[j];
    }
#pragma unroll
    for (int nt = 0; nt < 4; ++nt) {
      const bf16_t* bn = bc + nt * 16 * BK;   // row n = nt*16 + l15
      bh8 blo = *(const bh8*)(bn);
      bh8 bhi = *(const bh8*)(bn + 16);
      v16bf b;
#pragma unroll
      for (int j = 0; j < 8; ++j) { b[j] = blo[j]; b[j + 8] = bhi[j]; }
      acc[0][nt] = __builtin_amdgcn_wmma_f32_16x16x32_bf16(
          false, a0, false, b, (short)0, acc[0][nt], false, false);
      acc[1][nt] = __builtin_amdgcn_wmma_f32_16x16x32_bf16(
          false, a1, false, b, (short)0, acc[1][nt], false, false);
    }
  }
}

// Generic GEMM -> bf16 output, optional bias. Batched via blockIdx.z + strides.
template<bool BT>
__global__ __launch_bounds__(256)
void gemm_bf16_kernel(const bf16_t* __restrict__ A, const bf16_t* __restrict__ Bmat,
                      const float* __restrict__ bias, bf16_t* __restrict__ Out,
                      int lda, int ldb, int ldo, int K,
                      long long sA, long long sB, long long sO) {
  __shared__ GemmSmem sm;
  const int tileN = blockIdx.x * BN;
  const int tileM = blockIdx.y * BM;
  const long long bz = blockIdx.z;
  const bf16_t* Ap = A + bz * sA + (long long)tileM * lda;
  const bf16_t* Bp = BT ? (Bmat + bz * sB + (long long)tileN * ldb)
                        : (Bmat + bz * sB + tileN);
  v8f acc[2][4]; zero_acc(acc);
  gemm_core<BT>(Ap, lda, Bp, ldb, K, &sm, acc);

  const int lane = threadIdx.x & 31, wave = threadIdx.x >> 5;
  const int ncol = lane & 15;
#pragma unroll
  for (int mt = 0; mt < 2; ++mt) {
    const int mbase = tileM + wave * 32 + mt * 16 + ((lane >> 4) << 3);
#pragma unroll
    for (int nt = 0; nt < 4; ++nt) {
      int n = tileN + nt * 16 + ncol;
      float bv = bias ? bias[n] : 0.0f;
#pragma unroll
      for (int r = 0; r < 8; ++r) {
        long long m = mbase + r;
        Out[bz * sO + m * ldo + n] = f32_to_bf16(acc[mt][nt][r] + bv);
      }
    }
  }
}

// Scores: S[b,q,k] = A[b,q,:]·C[b,k,:] + u[b,q] + v[b,k] + keypad[b,k]  (f32 out)
__global__ __launch_bounds__(256)
void score_kernel(const bf16_t* __restrict__ Abf, const bf16_t* __restrict__ Cbf,
                  const float* __restrict__ u, const float* __restrict__ v,
                  const float* __restrict__ kp, float* __restrict__ S) {
  __shared__ GemmSmem sm;
  const int tileN = blockIdx.x * BN;   // keys
  const int tileM = blockIdx.y * BM;   // queries
  const long long b = blockIdx.z;
  v8f acc[2][4]; zero_acc(acc);
  gemm_core<true>(Abf + (b * LSEQ + tileM) * (long long)DIM, DIM,
                  Cbf + (b * LSEQ + tileN) * (long long)DIM, DIM, DIM, &sm, acc);

  const int lane = threadIdx.x & 31, wave = threadIdx.x >> 5;
#pragma unroll
  for (int mt = 0; mt < 2; ++mt) {
    const int qbase = tileM + wave * 32 + mt * 16 + ((lane >> 4) << 3);
#pragma unroll
    for (int nt = 0; nt < 4; ++nt) {
      int k = tileN + nt * 16 + (lane & 15);
      float vk = v[b * LSEQ + k] + kp[b * LSEQ + k];
#pragma unroll
      for (int r = 0; r < 8; ++r) {
        long long q = qbase + r;
        S[(b * LSEQ + q) * (long long)LSEQ + k] = acc[mt][nt][r] + u[b * LSEQ + q] + vk;
      }
    }
  }
}

// Final: tanh([output | mix] @ Wf^T + bf), masked, atomic-accumulated into pool[b,:]
__global__ __launch_bounds__(256)
void final_kernel(const bf16_t* __restrict__ out_bf, const bf16_t* __restrict__ mix_bf,
                  const bf16_t* __restrict__ Wf_bf, const float* __restrict__ bf_bias,
                  const int* __restrict__ mask, float* __restrict__ pool) {
  __shared__ GemmSmem sm;
  const int tileN = blockIdx.x * BN;
  const int tileM = blockIdx.y * BM;          // global row in [0, B*L)
  v8f acc[2][4]; zero_acc(acc);
  // split-K: k in [0,D) from `output`, k in [D,2D) from `mix`
  gemm_core<true>(out_bf + (long long)tileM * DIM, DIM,
                  Wf_bf + (long long)tileN * (2 * DIM), 2 * DIM, DIM, &sm, acc);
  gemm_core<true>(mix_bf + (long long)tileM * DIM, DIM,
                  Wf_bf + (long long)tileN * (2 * DIM) + DIM, 2 * DIM, DIM, &sm, acc);

  const int lane = threadIdx.x & 31, wave = threadIdx.x >> 5;
  const int bidx = tileM / LSEQ;              // whole tile lies in one batch (L % BM == 0)
#pragma unroll
  for (int mt = 0; mt < 2; ++mt) {
    const int rbase = tileM + wave * 32 + mt * 16 + ((lane >> 4) << 3);
#pragma unroll
    for (int nt = 0; nt < 4; ++nt) {
      int n = tileN + nt * 16 + (lane & 15);
      float bb = bf_bias[n];
#pragma unroll
      for (int r = 0; r < 8; ++r) {
        int gr = rbase + r;                   // flat (b,l) row; mask is flat [B*L]
        float mv = (float)mask[gr];
        float val = tanhf(acc[mt][nt][r] + bb) * mv;
        atomicAdd(&pool[(long long)bidx * DIM + n], val);
      }
    }
  }
}

// -------------------------- small helper kernels --------------------------

__device__ __forceinline__ float wave_sum(float v) {
#pragma unroll
  for (int o = 16; o > 0; o >>= 1) v += __shfl_down(v, o, 32);
  return v;
}
__device__ __forceinline__ float wave_max(float v) {
#pragma unroll
  for (int o = 16; o > 0; o >>= 1) v = fmaxf(v, __shfl_down(v, o, 32));
  return v;
}

// one block (256 threads) per row of S; P = softmax(S) as bf16
__global__ __launch_bounds__(256)
void softmax_kernel(const float* __restrict__ S, bf16_t* __restrict__ P) {
  __shared__ float red_m[8];
  __shared__ float red_s[8];
  const long long row = blockIdx.x;
  const float* s = S + row * LSEQ;
  bf16_t* p = P + row * LSEQ;
  const int t = threadIdx.x, lane = t & 31, wave = t >> 5;

  float vals[4]; float mx = -3.4e38f;
#pragma unroll
  for (int i = 0; i < 4; ++i) { vals[i] = s[t + i * 256]; mx = fmaxf(mx, vals[i]); }
  mx = wave_max(mx);
  if (lane == 0) red_m[wave] = mx;
  __syncthreads();
  float M = red_m[0];
#pragma unroll
  for (int i = 1; i < 8; ++i) M = fmaxf(M, red_m[i]);

  float ssum = 0.0f;
#pragma unroll
  for (int i = 0; i < 4; ++i) { vals[i] = __expf(vals[i] - M); ssum += vals[i]; }
  ssum = wave_sum(ssum);
  if (lane == 0) red_s[wave] = ssum;
  __syncthreads();
  float T = 0.0f;
#pragma unroll
  for (int i = 0; i < 8; ++i) T += red_s[i];
  float inv = 1.0f / T;
#pragma unroll
  for (int i = 0; i < 4; ++i) p[t + i * 256] = f32_to_bf16(vals[i] * inv);
}

// out[row] = dot(X[row,:], w) + bias[0]; one 256-thread block per row
__global__ __launch_bounds__(256)
void rowdot_kernel(const float* __restrict__ X, const float* __restrict__ w,
                   const float* __restrict__ bias, float* __restrict__ out) {
  __shared__ float red[8];
  const long long row = blockIdx.x;
  const float* x = X + row * DIM;
  float s = 0.0f;
  for (int i = threadIdx.x; i < DIM; i += 256) s += x[i] * w[i];
  s = wave_sum(s);
  const int lane = threadIdx.x & 31, wave = threadIdx.x >> 5;
  if (lane == 0) red[wave] = s;
  __syncthreads();
  if (threadIdx.x == 0) {
    float tot = 0.0f;
#pragma unroll
    for (int i = 0; i < 8; ++i) tot += red[i];
    out[row] = tot + bias[0];
  }
}

// per-batch: lengths[b] = sum(mask), keypad = mask ? 0 : -1e9
__global__ __launch_bounds__(256)
void mask_kernel(const int* __restrict__ mask, float* __restrict__ keypad,
                 float* __restrict__ lengths) {
  __shared__ float red[8];
  const int b = blockIdx.x;
  float s = 0.0f;
  for (int i = threadIdx.x; i < LSEQ; i += 256) {
    int mv = mask[b * LSEQ + i];
    keypad[b * LSEQ + i] = (mv > 0) ? 0.0f : NEGC;
    s += (float)mv;
  }
  s = wave_sum(s);
  const int lane = threadIdx.x & 31, wave = threadIdx.x >> 5;
  if (lane == 0) red[wave] = s;
  __syncthreads();
  if (threadIdx.x == 0) {
    float tot = 0.0f;
#pragma unroll
    for (int i = 0; i < 8; ++i) tot += red[i];
    lengths[b] = tot;
  }
}

__global__ __launch_bounds__(256)
void cvt_bf16_kernel(const float* __restrict__ src, bf16_t* __restrict__ dst, long long n) {
  long long i = (long long)blockIdx.x * 256 + threadIdx.x;
  long long stride = (long long)gridDim.x * 256;
  for (; i < n; i += stride) dst[i] = f32_to_bf16(src[i]);
}

// bbo[n] = sum_d Wb[n,d] * bo[d]   (bias carried through the fused Wbo GEMM)
__global__ __launch_bounds__(256)
void bbo_kernel(const float* __restrict__ Wb, const float* __restrict__ bo,
                float* __restrict__ bbo) {
  int n = blockIdx.x * 256 + threadIdx.x;
  if (n >= DIM) return;
  float s = 0.0f;
  for (int d = 0; d < DIM; ++d) s += Wb[(long long)n * DIM + d] * bo[d];
  bbo[n] = s;
}

__global__ __launch_bounds__(256)
void zero_kernel(float* __restrict__ p, long long n) {
  long long i = (long long)blockIdx.x * 256 + threadIdx.x;
  if (i < n) p[i] = 0.0f;
}

__global__ __launch_bounds__(256)
void pool_final_kernel(const float* __restrict__ pool, const float* __restrict__ lengths,
                       float* __restrict__ out) {
  const int b = blockIdx.x;
  float inv = 1.0f / lengths[b];
  for (int i = threadIdx.x; i < DIM; i += 256)
    out[(long long)b * DIM + i] = pool[(long long)b * DIM + i] * inv;
}

// ================================ launcher ================================

extern "C" void kernel_launch(void* const* d_in, const int* in_sizes, int n_in,
                              void* d_out, int out_size, void* d_ws, size_t ws_size,
                              hipStream_t stream) {
  (void)in_sizes; (void)n_in; (void)out_size; (void)ws_size;
  const float* output  = (const float*)d_in[0];
  const float* context = (const float*)d_in[1];
  const int*   mask    = (const int*)d_in[2];
  const float* Wo      = (const float*)d_in[3];
  const float* bo      = (const float*)d_in[4];
  const float* Wc      = (const float*)d_in[5];
  const float* bc      = (const float*)d_in[6];
  const float* Wb      = (const float*)d_in[7];
  const float* Wu      = (const float*)d_in[8];
  const float* bu      = (const float*)d_in[9];
  const float* Wv      = (const float*)d_in[10];
  const float* bv      = (const float*)d_in[11];
  const float* Wf      = (const float*)d_in[12];
  const float* bfb     = (const float*)d_in[13];
  float* dout = (float*)d_out;

  const long long NLD = (long long)BATCH * LSEQ * DIM;   // 33,554,432
  const long long NLL = (long long)BATCH * LSEQ * LSEQ;  // 33,554,432
  const long long DD  = (long long)DIM * DIM;

  size_t off = 0;
  auto carve = [&](size_t bytes) -> void* {
    void* r = (char*)d_ws + off;
    off = (off + bytes + 255) & ~(size_t)255;
    return r;
  };
  bf16_t* out_bf  = (bf16_t*)carve(NLD * 2);
  bf16_t* ctx_bf  = (bf16_t*)carve(NLD * 2);
  bf16_t* A_bf    = (bf16_t*)carve(NLD * 2);
  bf16_t* C_bf    = (bf16_t*)carve(NLD * 2);
  bf16_t* mix_bf  = (bf16_t*)carve(NLD * 2);
  float*  S       = (float*) carve(NLL * 4);
  bf16_t* P_bf    = (bf16_t*)carve(NLL * 2);
  bf16_t* Wo_bf   = (bf16_t*)carve(DD * 2);
  bf16_t* Wb_bf   = (bf16_t*)carve(DD * 2);
  bf16_t* Wc_bf   = (bf16_t*)carve(DD * 2);
  bf16_t* Wbo_bf  = (bf16_t*)carve(DD * 2);
  bf16_t* Wf_bf   = (bf16_t*)carve(DD * 4);  // [D, 2D]
  float*  u       = (float*) carve((long long)BATCH * LSEQ * 4);
  float*  v       = (float*) carve((long long)BATCH * LSEQ * 4);
  float*  keypad  = (float*) carve((long long)BATCH * LSEQ * 4);
  float*  lengths = (float*) carve(BATCH * 4);
  float*  bbo     = (float*) carve(DIM * 4);
  float*  pool    = (float*) carve((long long)BATCH * DIM * 4);

  const dim3 blk(256);

  // 0) zero the pooled accumulator (d_ws is poisoned, never auto-cleared)
  zero_kernel<<<dim3((BATCH * DIM + 255) / 256), blk, 0, stream>>>(pool, (long long)BATCH * DIM);

  // 1) f32 -> bf16 conversions
  cvt_bf16_kernel<<<dim3(2048), blk, 0, stream>>>(output,  out_bf, NLD);
  cvt_bf16_kernel<<<dim3(2048), blk, 0, stream>>>(context, ctx_bf, NLD);
  cvt_bf16_kernel<<<dim3(512),  blk, 0, stream>>>(Wo, Wo_bf, DD);
  cvt_bf16_kernel<<<dim3(512),  blk, 0, stream>>>(Wb, Wb_bf, DD);
  cvt_bf16_kernel<<<dim3(512),  blk, 0, stream>>>(Wc, Wc_bf, DD);
  cvt_bf16_kernel<<<dim3(1024), blk, 0, stream>>>(Wf, Wf_bf, DD * 2);

  // 2) mask stats, biaffine rank-1 terms, fused bilinear bias
  mask_kernel<<<dim3(BATCH), blk, 0, stream>>>(mask, keypad, lengths);
  rowdot_kernel<<<dim3(BATCH * LSEQ), blk, 0, stream>>>(output,  Wu, bu, u);
  rowdot_kernel<<<dim3(BATCH * LSEQ), blk, 0, stream>>>(context, Wv, bv, v);
  bbo_kernel<<<dim3((DIM + 255) / 256), blk, 0, stream>>>(Wb, bo, bbo);

  // 3) Wbo = Wb @ Wo  (fuses output_trans + bilinear into one projection)
  gemm_bf16_kernel<false><<<dim3(DIM / BN, DIM / BM, 1), blk, 0, stream>>>(
      Wb_bf, Wo_bf, nullptr, Wbo_bf, DIM, DIM, DIM, DIM, 0, 0, 0);

  // 4) A = output @ Wbo^T + Wb·bo ; C = context @ Wc^T + bc
  gemm_bf16_kernel<true><<<dim3(DIM / BN, (BATCH * LSEQ) / BM, 1), blk, 0, stream>>>(
      out_bf, Wbo_bf, bbo, A_bf, DIM, DIM, DIM, DIM, 0, 0, 0);
  gemm_bf16_kernel<true><<<dim3(DIM / BN, (BATCH * LSEQ) / BM, 1), blk, 0, stream>>>(
      ctx_bf, Wc_bf, bc, C_bf, DIM, DIM, DIM, DIM, 0, 0, 0);

  // 5) scores S = A @ C^T + u + v + keypad  (batched over B)
  score_kernel<<<dim3(LSEQ / BN, LSEQ / BM, BATCH), blk, 0, stream>>>(
      A_bf, C_bf, u, v, keypad, S);

  // 6) row softmax -> bf16 probabilities
  softmax_kernel<<<dim3(BATCH * LSEQ), blk, 0, stream>>>(S, P_bf);

  // 7) mix = P @ context  (batched, non-transposed B)
  gemm_bf16_kernel<false><<<dim3(DIM / BN, LSEQ / BM, BATCH), blk, 0, stream>>>(
      P_bf, ctx_bf, nullptr, mix_bf, LSEQ, DIM, DIM, LSEQ,
      (long long)LSEQ * LSEQ, (long long)LSEQ * DIM, (long long)LSEQ * DIM);

  // 8) final: tanh([output|mix] @ Wf^T + bf), masked atomic pooling
  final_kernel<<<dim3(DIM / BN, (BATCH * LSEQ) / BM, 1), blk, 0, stream>>>(
      out_bf, mix_bf, Wf_bf, bfb, mask, pool);

  // 9) pooled mean
  pool_final_kernel<<<dim3(BATCH), blk, 0, stream>>>(pool, lengths, dout);
}
